// _gcn_27745488732942
// MI455X (gfx1250) — matrix operand order repaired and checked by reading, not compiled, not run on hardware
//
#include <hip/hip_runtime.h>
#include <math.h>

// ---------------------------------------------------------------------------
// Plain ext-vector POD types (safe in unions)
// ---------------------------------------------------------------------------
typedef __attribute__((ext_vector_type(16))) __bf16 v16bf;
typedef __attribute__((ext_vector_type(8)))  float  v8f;
typedef __attribute__((ext_vector_type(4)))  unsigned int u32x4;
typedef __attribute__((ext_vector_type(8)))  int    i32x8;
typedef __attribute__((ext_vector_type(4)))  int    i32x4;
typedef __attribute__((ext_vector_type(4)))  float  f32x4;

union FragBF {
    v16bf v;
    unsigned short us[16];
    unsigned int   d[8];
    u32x4 q[2];
};

__device__ __forceinline__ unsigned fbits(float f) {
    union { float f; unsigned u; } c; c.f = f; return c.u;
}
// fp32 -> bf16 (round-to-nearest-even) for one scalar (prep kernels)
__device__ __forceinline__ unsigned short f2bf(float f) {
    unsigned u = fbits(f);
    return (unsigned short)((u + 0x7FFFu + ((u >> 16) & 1u)) >> 16);
}
// pack two fp32 -> two bf16 (truncate) in ONE v_perm_b32
__device__ __forceinline__ unsigned pack2bf(float lo, float hi) {
    return __builtin_amdgcn_perm(fbits(hi), fbits(lo), 0x07060302u);
}

#define CIN 4096
#define DM  512
#define NT  512
#define MM  12

#if defined(__gfx1250__) && __has_builtin(__builtin_amdgcn_tensor_load_to_lds) && \
    __has_builtin(__builtin_amdgcn_s_wait_tensorcnt)
#define HAVE_TDM 1
#else
#define HAVE_TDM 0
#endif

// ---------------------------------------------------------------------------
// Kernel 0: prep.
//   wa_bf  [1024][4096] : bf16 copy of w_a (halves GEMM A-side bytes, kills cvt VALU)
//   wcat_bf[2048][1024] : bf16 [w_ih | w_hh] rows
//   bsum, zero LSTM state.
// Grid 16384 x 256  (4M threads)
// ---------------------------------------------------------------------------
__global__ void prep_kernel(const float* __restrict__ w_a,
                            const float* __restrict__ w_ih,
                            const float* __restrict__ w_hh,
                            const float* __restrict__ b_ih,
                            const float* __restrict__ b_hh,
                            unsigned short* __restrict__ wa_bf,
                            unsigned short* __restrict__ wcat_bf,
                            float* __restrict__ bsum,
                            float* __restrict__ cstate,
                            unsigned short* __restrict__ h_bf) {
    int idx = blockIdx.x * blockDim.x + threadIdx.x;
    if (idx < 1024 * CIN) wa_bf[idx] = f2bf(w_a[idx]);
    if (idx < 2048 * 1024) {
        int j = idx >> 10;
        int d = idx & 1023;
        float v = (d < 512) ? w_ih[j * 512 + d] : w_hh[j * 512 + (d - 512)];
        wcat_bf[idx] = f2bf(v);
    }
    if (idx < 2048) bsum[idx] = b_ih[idx] + b_hh[idx];
    if (idx < 32 * 512) { cstate[idx] = 0.0f; h_bf[idx] = 0; }
}

// ---------------------------------------------------------------------------
// Kernel 1: node1[n][o][m] = sum_c base_out[n,m,c] * w_a[o,c]
//
// Block = (o half, n): 8 waves x 4 o-tiles = 512 o rows, one n.
// B operand (base_out rows of this n) staged in LDS:
//   TDM path : tensor_load_to_lds, 2D tensor (4096 x 12), tile (512 x 12),
//              double buffered 2 x 24KB, s_wait_tensorcnt + barrier.
//   fallback : cooperative global->LDS copy.
// Each wave: 4 f32 accumulators; per 32-K step: 1 B frag (LDS, v_perm cvt)
// shared by 4 WMMAs, 4 bf16 A frags loaded straight from wa_bf.
// Grid (2, 512), 256 threads.
// ---------------------------------------------------------------------------
__global__ void gemm_node1_kernel(const float* __restrict__ base_out,
                                  const unsigned short* __restrict__ wa_bf,
                                  float* __restrict__ node1) {
    __shared__ float ldsB[2][MM * 512];   // 2 x 24KB

    const int wave = threadIdx.x >> 5;
    const int lane = threadIdx.x & 31;
    const int n    = blockIdx.y;
    const int o_base = blockIdx.x * 512 + wave * 64;
    const int col  = lane & 15;           // A row in tile / D col (m)
    const int kg   = lane >> 4;           // lane half
    const int vb   = (col < MM) ? col : (MM - 1);

    const float* __restrict__ gB = base_out + (size_t)n * MM * CIN;
    const unsigned short* __restrict__ arow0 = wa_bf + (size_t)(o_base + col) * CIN;

    v8f acc[4] = {v8f{}, v8f{}, v8f{}, v8f{}};

    // ---- one 512-K slab of B from LDS buffer lB, K offset kb ----
    auto compute_slab = [&](const float* lB, int kb) {
        for (int kl = 0; kl < 512; kl += 32) {
            // B fragment: col m = vb, K = kb+kl + 16*kg + [0..15] contiguous fp32
            const f32x4* pb = (const f32x4*)(lB + vb * 512 + kl + 16 * kg);
            f32x4 b0 = pb[0], b1 = pb[1], b2 = pb[2], b3 = pb[3];
            float bx[16];
#pragma unroll
            for (int i = 0; i < 4; ++i) {
                bx[i] = b0[i]; bx[4 + i] = b1[i]; bx[8 + i] = b2[i]; bx[12 + i] = b3[i];
            }
            FragBF bfr;
#pragma unroll
            for (int i = 0; i < 8; ++i)
                bfr.d[i] = pack2bf(bx[2 * i], bx[2 * i + 1]);

#pragma unroll
            for (int t = 0; t < 4; ++t) {
                FragBF af;
                const unsigned short* ar =
                    arow0 + (size_t)t * 16 * CIN + kb + kl + 16 * kg;
                af.q[0] = *(const u32x4*)ar;         // K = +16*kg + [0..7]
                af.q[1] = *(const u32x4*)(ar + 8);   // K = +16*kg + [8..15]
                acc[t] = __builtin_amdgcn_wmma_f32_16x16x32_bf16(
                    false, af.v, false, bfr.v, (short)0, acc[t], false, false);
            }
        }
    };

#if HAVE_TDM
    // ---- Tensor Data Mover producer, double buffered ----
    auto issue_tdm = [&](int buf, int kb) {
        unsigned long long ga =
            (unsigned long long)(const char*)gB + (unsigned long long)kb * 4ull;
        u32x4 g0;
        g0[0] = 1u;                                   // count=1, user mode
        g0[1] = (unsigned)(buf * (MM * 512 * 4));     // lds_addr (bytes)
        g0[2] = (unsigned)(ga & 0xFFFFFFFFu);         // global_addr[31:0]
        g0[3] = (unsigned)((ga >> 32) & 0x1FFFFFFu) | (2u << 30); // addr[56:32] | type=2
        i32x8 g1;
        g1[0] = (int)(2u << 16);          // data_size = 4B
        g1[1] = (int)(4096u << 16);       // tensor_dim0[15:0] = 4096
        g1[2] = (int)(12u << 16);         // tensor_dim0[31:16]=0 | tensor_dim1 = 12
        g1[3] = (int)(512u << 16);        // tensor_dim1 hi=0 | tile_dim0 = 512
        g1[4] = 12;                       // tile_dim1 = 12 (tile_dim2 = 0)
        g1[5] = 4096;                     // tensor_dim0_stride = 4096
        g1[6] = 0;
        g1[7] = 0;
        i32x4 gz4 = {0, 0, 0, 0};
        i32x8 gz8 = {0, 0, 0, 0, 0, 0, 0, 0};
        // clang-23 / amdgpu-toolchain 6-arg form (probe: groups 0..4 + cpol)
        __builtin_amdgcn_tensor_load_to_lds(g0, g1, gz4, gz4, gz8, 0);
    };

    if (threadIdx.x == 0) issue_tdm(0, 0);            // prologue
    __builtin_amdgcn_s_wait_tensorcnt((short)0);
    __syncthreads();

    for (int kb = 0; kb < CIN; kb += 512) {
        int buf = (kb >> 9) & 1;
        if (threadIdx.x == 0 && (kb + 512) < CIN)
            issue_tdm(buf ^ 1, kb + 512);             // overlap DMA with compute
        compute_slab(ldsB[buf], kb);
        __builtin_amdgcn_s_wait_tensorcnt((short)0);  // next buffer landed
        __syncthreads();
    }
#else
    // ---- fallback: cooperative staging, single buffer ----
    for (int kb = 0; kb < CIN; kb += 512) {
        __syncthreads();
        for (int idx = threadIdx.x; idx < MM * 512; idx += 256)
            ldsB[0][idx] = gB[(idx >> 9) * CIN + kb + (idx & 511)];
        __syncthreads();
        compute_slab(ldsB[0], kb);
    }
#endif

    // D layout: col = lane&15, row = r + 8*kg
    if (col < MM) {
#pragma unroll
        for (int t = 0; t < 4; ++t) {
            int o0 = o_base + t * 16;
#pragma unroll
            for (int r = 0; r < 8; ++r)
                node1[((size_t)n * 1024 + o0 + r + 8 * kg) * MM + col] = acc[t][r];
        }
    }
}

// ---------------------------------------------------------------------------
// Kernel 2: A = sum of 3 dist slices; node[n,c,w] = sum_{k,v} node1 * A;
// relu; mean over w; store bf16 seq row.  One block per n.
// ---------------------------------------------------------------------------
__global__ void graph_pool_kernel(const float* __restrict__ node1,
                                  const float* __restrict__ dist,
                                  unsigned short* __restrict__ seq_bf) {
    __shared__ float As[2][MM][MM];
    const int n = blockIdx.x;
    for (int t = threadIdx.x; t < 2 * MM * MM; t += blockDim.x) {
        int k = t / (MM * MM);
        int vw = t % (MM * MM);
        float s = 0.0f;
#pragma unroll
        for (int sl = 0; sl < 3; ++sl)
            s += dist[((((size_t)n * 3 + sl) * 2 + k) * (MM * MM)) + vw];
        As[k][vw / MM][vw % MM] = s;
    }
    __syncthreads();

    for (int c = threadIdx.x; c < DM; c += blockDim.x) {
        float accw[MM];
#pragma unroll
        for (int w = 0; w < MM; ++w) accw[w] = 0.0f;
        for (int k = 0; k < 2; ++k) {
            const float* nrow = node1 + ((size_t)n * 1024 + k * DM + c) * MM;
#pragma unroll
            for (int v = 0; v < MM; ++v) {
                float x = nrow[v];
#pragma unroll
                for (int w = 0; w < MM; ++w) accw[w] += x * As[k][v][w];
            }
        }
        float m = 0.0f;
#pragma unroll
        for (int w = 0; w < MM; ++w) m += fmaxf(accw[w], 0.0f);
        seq_bf[(size_t)n * DM + c] = f2bf(m * (1.0f / 12.0f));
    }
}

// ---------------------------------------------------------------------------
// Kernel 3 (x16): LSTM gate pre-activations for step s:
//   g[b][j] = [x_s | h] @ wcat[j][:]  + (b_ih[j]+b_hh[j])
// Output 32x2048 -> 256 tiles of 16x16; 32 blocks x 8 waves; K = 1024 by 32.
// All operands already bf16 -> zero conversion VALU in the loop.
// ---------------------------------------------------------------------------
__global__ void lstm_gemm_kernel(const unsigned short* __restrict__ seq_bf,
                                 const unsigned short* __restrict__ h_bf,
                                 const unsigned short* __restrict__ wcat_bf,
                                 const float* __restrict__ bsum,
                                 float* __restrict__ g, int step) {
    const int wave = threadIdx.x >> 5;
    const int lane = threadIdx.x & 31;
    const int tile = blockIdx.x * 8 + wave;      // 0..255
    const int btile = tile >> 7;                 // 0..1   (batch rows)
    const int jtile = tile & 127;                // 0..127 (gate cols)
    const int col = lane & 15;
    const int kg  = lane >> 4;

    const int b = btile * 16 + col;
    const unsigned short* __restrict__ xrow = seq_bf + ((size_t)step * 32 + b) * DM;
    const unsigned short* __restrict__ hrow = h_bf + (size_t)b * DM;
    const unsigned short* __restrict__ wrow = wcat_bf + (size_t)(jtile * 16 + col) * 1024;

    v8f acc = {};
    for (int kb = 0; kb < 1024; kb += 32) {
        const unsigned short* arow = (kb < DM) ? (xrow + kb) : (hrow + (kb - DM));
        FragBF af, bfr;
        af.q[0]  = *(const u32x4*)(arow + 16 * kg);
        af.q[1]  = *(const u32x4*)(arow + 16 * kg + 8);
        bfr.q[0] = *(const u32x4*)(wrow + kb + 16 * kg);
        bfr.q[1] = *(const u32x4*)(wrow + kb + 16 * kg + 8);
        acc = __builtin_amdgcn_wmma_f32_16x16x32_bf16(
            false, af.v, false, bfr.v, (short)0, acc, false, false);
    }

    const int j = jtile * 16 + col;
    const float bias = bsum[j];
#pragma unroll
    for (int r = 0; r < 8; ++r) {
        int bb = btile * 16 + r + 8 * kg;
        g[(size_t)bb * 2048 + j] = acc[r] + bias;
    }
}

// ---------------------------------------------------------------------------
// Kernel 4 (x16): gate nonlinearity + state update. Saves h[b==31] per step.
// ---------------------------------------------------------------------------
__global__ void lstm_gates_kernel(const float* __restrict__ g,
                                  float* __restrict__ cstate,
                                  unsigned short* __restrict__ h_bf,
                                  float* __restrict__ hs_last, int step) {
    int idx = blockIdx.x * blockDim.x + threadIdx.x;
    if (idx >= 32 * DM) return;
    int b = idx >> 9;
    int d = idx & (DM - 1);
    const float* gr = g + (size_t)b * 2048;
    float gi = gr[d], gf = gr[DM + d], gg = gr[2 * DM + d], go = gr[3 * DM + d];
    float si = 1.0f / (1.0f + __expf(-gi));
    float sf = 1.0f / (1.0f + __expf(-gf));
    float so = 1.0f / (1.0f + __expf(-go));
    float cc = sf * cstate[idx] + si * tanhf(gg);
    float hh = so * tanhf(cc);
    cstate[idx] = cc;
    h_bf[idx] = f2bf(hh);
    if (b == 31) hs_last[(size_t)step * DM + d] = hh;
}

// ---------------------------------------------------------------------------
// Kernel 5: out[s][cls] = hs_last[s] . w_cls[cls] + b_cls[cls]   (16x8)
// ---------------------------------------------------------------------------
__global__ void classify_kernel(const float* __restrict__ hs_last,
                                const float* __restrict__ w_cls,
                                const float* __restrict__ b_cls,
                                float* __restrict__ out) {
    int idx = threadIdx.x;
    if (idx >= 16 * 8) return;
    int s = idx >> 3, cl = idx & 7;
    const float* hr = hs_last + (size_t)s * DM;
    const float* wr = w_cls + (size_t)cl * DM;
    float acc = 0.0f;
    for (int d = 0; d < DM; ++d) acc += hr[d] * wr[d];
    out[idx] = acc + b_cls[cl];
}

// ---------------------------------------------------------------------------
extern "C" void kernel_launch(void* const* d_in, const int* in_sizes, int n_in,
                              void* d_out, int out_size, void* d_ws, size_t ws_size,
                              hipStream_t stream) {
    const float* base_out = (const float*)d_in[0];
    const float* dist     = (const float*)d_in[1];
    const float* w_a      = (const float*)d_in[2];
    const float* w_ih     = (const float*)d_in[3];
    const float* w_hh     = (const float*)d_in[4];
    const float* b_ih     = (const float*)d_in[5];
    const float* b_hh     = (const float*)d_in[6];
    const float* w_cls    = (const float*)d_in[7];
    const float* b_cls    = (const float*)d_in[8];
    float* out = (float*)d_out;

    char* ws = (char*)d_ws;
    float*          node1   = (float*)ws;          ws += (size_t)NT * 1024 * MM * 4;  // 25.2 MB
    unsigned short* seq_bf  = (unsigned short*)ws; ws += (size_t)NT * DM * 2;         // 512 KB
    unsigned short* wcat_bf = (unsigned short*)ws; ws += (size_t)2048 * 1024 * 2;     // 4 MB
    unsigned short* wa_bf   = (unsigned short*)ws; ws += (size_t)1024 * CIN * 2;      // 8 MB
    float*          bsum    = (float*)ws;          ws += 2048 * 4;
    float*          g       = (float*)ws;          ws += 32 * 2048 * 4;
    float*          cstate  = (float*)ws;          ws += 32 * DM * 4;
    unsigned short* h_bf    = (unsigned short*)ws; ws += 32 * DM * 2;
    float*          hs_last = (float*)ws;          ws += 16 * DM * 4;

    prep_kernel<<<16384, 256, 0, stream>>>(w_a, w_ih, w_hh, b_ih, b_hh,
                                           wa_bf, wcat_bf, bsum, cstate, h_bf);

    gemm_node1_kernel<<<dim3(2, 512), 256, 0, stream>>>(base_out, wa_bf, node1);

    graph_pool_kernel<<<NT, 256, 0, stream>>>(node1, dist, seq_bf);

    for (int s = 0; s < 16; ++s) {
        lstm_gemm_kernel<<<32, 256, 0, stream>>>(seq_bf, h_bf, wcat_bf, bsum, g, s);
        lstm_gates_kernel<<<64, 256, 0, stream>>>(g, cstate, h_bf, hs_last, s);
    }

    classify_kernel<<<1, 128, 0, stream>>>(hs_last, w_cls, b_cls, out);
}
